// FastSelfAttention_89343909691875
// MI455X (gfx1250) — matrix operand order repaired and checked by reading, not compiled, not run on hardware
//
#include <hip/hip_runtime.h>
#include <stdint.h>
#include <math.h>

// ---------------- problem constants ----------------
#define B_NUM 8
#define S_LEN 2048
#define D_DIM 1024
#define H_NUM 16
#define DH    64
#define ROWS  (B_NUM * S_LEN)   // 16384

typedef __attribute__((ext_vector_type(16))) __bf16 v16bf;
typedef __attribute__((ext_vector_type(8)))  float  v8f;

union Frag16 { v16bf v; uint4 q[2]; };

__device__ __forceinline__ unsigned short f2bf(float f) {
  unsigned int u = __float_as_uint(f);
  u += 0x7FFFu + ((u >> 16) & 1u);           // round-to-nearest-even
  return (unsigned short)(u >> 16);
}
__device__ __forceinline__ float bf2f(unsigned short h) {
  return __uint_as_float(((unsigned int)h) << 16);
}

// --- CDNA5 async global->LDS copy (ASYNCcnt path), 16 B per lane ---
__device__ __forceinline__ void async_copy_b128(unsigned lds_off, const void* gsrc) {
  asm volatile("global_load_async_to_lds_b128 %0, %1, off"
               :: "v"(lds_off), "v"(gsrc) : "memory");
}
#define ASYNC_WAIT(imm) asm volatile("s_wait_asynccnt " #imm ::: "memory")

__device__ __forceinline__ unsigned lds_off_of(const void* p) {
  // flat LDS address = {aperture, offset}; low 32 bits are the LDS byte offset
  return (unsigned)(uintptr_t)p;
}

// ---------------- pack kernels ----------------
__global__ void pack_bf16(const float* __restrict__ s, unsigned short* __restrict__ d, int n) {
  int i = (blockIdx.x * blockDim.x + threadIdx.x) * 4;
  if (i + 3 < n) {
    float4 v = *(const float4*)(s + i);
    d[i] = f2bf(v.x); d[i+1] = f2bf(v.y); d[i+2] = f2bf(v.z); d[i+3] = f2bf(v.w);
  } else {
    for (int j = i; j < n; ++j) d[j] = f2bf(s[j]);
  }
}

// f32 [R,C] -> bf16 [C,R]  (LDS-tiled transpose; block (32,8), grid (C/32, R/32))
__global__ void pack_transpose(const float* __restrict__ src, unsigned short* __restrict__ dst,
                               int R, int C) {
  __shared__ float tile[32][33];
  int c0 = blockIdx.x * 32, r0 = blockIdx.y * 32;
  int tx = threadIdx.x, ty = threadIdx.y;
  #pragma unroll
  for (int j = ty; j < 32; j += 8)
    tile[j][tx] = src[(size_t)(r0 + j) * C + c0 + tx];
  __syncthreads();
  #pragma unroll
  for (int j = ty; j < 32; j += 8)
    dst[(size_t)(c0 + j) * R + r0 + tx] = f2bf(tile[tx][j]);
}

// Wqa [D,H] f32 -> WqaT [H,D] bf16
__global__ void pack_wqaT(const float* __restrict__ wqa, unsigned short* __restrict__ dst) {
  int i = blockIdx.x * blockDim.x + threadIdx.x;   // 0 .. D*H-1
  int k = i >> 4, h = i & 15;
  dst[h * D_DIM + k] = f2bf(wqa[i]);
}

// weighted_bf[b,s,d] = bf16( q_f32[b,s,d] * pooledK[b,d] )   (4 elems/thread)
__global__ void scale_q(const float* __restrict__ Q, const float* __restrict__ pooledK,
                        unsigned short* __restrict__ Wbf) {
  int i = (blockIdx.x * blockDim.x + threadIdx.x) * 4;
  int b = i >> 21;                 // / (S_LEN*D_DIM)
  int d = i & (D_DIM - 1);
  float4 v = *(const float4*)(Q + i);
  const float* pk = pooledK + b * D_DIM + d;
  Wbf[i]     = f2bf(v.x * pk[0]);
  Wbf[i + 1] = f2bf(v.y * pk[1]);
  Wbf[i + 2] = f2bf(v.z * pk[2]);
  Wbf[i + 3] = f2bf(v.w * pk[3]);
}

// ---------------- main WMMA GEMM (async double-buffered) ----------------
// C[m,n] = sum_k A_bf16[m,k] * WT_bf16[n,k]  + bias[n] (+resid[m,n]); optional bf16 copy.
// Block: 256 thr (8 waves, 2x4), tile 128x256, per-wave 64x64, K-step 32.
#define LDA 40                    // padded LDS stride in halfs (80 B, b128-aligned)
#define ABUF (128 * LDA)          // halfs per A buffer
#define BBUF (256 * LDA)          // halfs per B buffer
__global__ void __launch_bounds__(256)
gemm_bf16_wmma(const unsigned short* __restrict__ A,
               const unsigned short* __restrict__ WT,   // n-major [N,K]
               const float* __restrict__ bias,
               const float* __restrict__ resid,
               float* __restrict__ C,
               unsigned short* __restrict__ Cbf)
{
  __shared__ unsigned short lsA[2 * ABUF];
  __shared__ unsigned short lsB[2 * BBUF];

  const int tid  = threadIdx.x;
  const int lane = tid & 31, wave = tid >> 5;
  const int wm = wave >> 2, wn = wave & 3;             // 2x4 wave grid
  const int m0 = blockIdx.y * 128, n0 = blockIdx.x * 256;
  const int r = lane & 15, half = lane >> 4;

  v8f acc[4][4];
  #pragma unroll
  for (int i = 0; i < 4; ++i)
    #pragma unroll
    for (int j = 0; j < 4; ++j)
      #pragma unroll
      for (int e = 0; e < 8; ++e) acc[i][j][e] = 0.f;

  // async stage one K-slice (A: 128x32, B: 256x32) into buffer `buf`
  auto issueGroup = [&](int buf, int k0) {
    #pragma unroll
    for (int i = 0; i < 2; ++i) {                       // A: 512 x 16B chunks
      int c = tid + i * 256;
      int row = c >> 2, piece = (c & 3) * 8;
      async_copy_b128(lds_off_of(&lsA[buf * ABUF + row * LDA + piece]),
                      A + (size_t)(m0 + row) * D_DIM + k0 + piece);
    }
    #pragma unroll
    for (int i = 0; i < 4; ++i) {                       // B: 1024 x 16B chunks
      int c = tid + i * 256;
      int n = c >> 2, piece = (c & 3) * 8;
      async_copy_b128(lds_off_of(&lsB[buf * BBUF + n * LDA + piece]),
                      WT + (size_t)(n0 + n) * D_DIM + k0 + piece);
    }
  };

  issueGroup(0, 0);
  int buf = 0;
  for (int k0 = 0; k0 < D_DIM; k0 += 32) {
    if (k0 + 32 < D_DIM) {
      issueGroup(buf ^ 1, k0 + 32);
      ASYNC_WAIT(6);          // 6 new ops outstanding -> current buffer complete
    } else {
      ASYNC_WAIT(0);
    }
    __syncthreads();

    Frag16 af[4], bfr[4];
    #pragma unroll
    for (int i = 0; i < 4; ++i) {
      const unsigned short* p = &lsA[buf * ABUF + (wm * 64 + i * 16 + r) * LDA];
      af[i].q[0] = *(const uint4*)(p + half * 8);       // K 0..7 / 8..15
      af[i].q[1] = *(const uint4*)(p + half * 8 + 16);  // K 16..23 / 24..31
    }
    #pragma unroll
    for (int j = 0; j < 4; ++j) {
      const unsigned short* p = &lsB[buf * BBUF + (wn * 64 + j * 16 + r) * LDA];
      bfr[j].q[0] = *(const uint4*)(p + half * 8);
      bfr[j].q[1] = *(const uint4*)(p + half * 8 + 16);
    }
    #pragma unroll
    for (int i = 0; i < 4; ++i)
      #pragma unroll
      for (int j = 0; j < 4; ++j)
        acc[i][j] = __builtin_amdgcn_wmma_f32_16x16x32_bf16(
            false, af[i].v, false, bfr[j].v, (short)0, acc[i][j], false, false);

    __syncthreads();
    buf ^= 1;
  }

  // epilogue: C/D layout (VGPR v, lane half): M = v + 8*half, N = r
  #pragma unroll
  for (int i = 0; i < 4; ++i) {
    int rowBase = m0 + wm * 64 + i * 16 + half * 8;
    #pragma unroll
    for (int j = 0; j < 4; ++j) {
      int col = n0 + wn * 64 + j * 16 + r;
      float bval = bias[col];
      #pragma unroll
      for (int v = 0; v < 8; ++v) {
        size_t idx = (size_t)(rowBase + v) * D_DIM + col;
        float val = acc[i][j][v] + bval;
        if (resid) val += resid[idx];
        C[idx] = val;
        if (Cbf) Cbf[idx] = f2bf(val);
      }
    }
  }
}

// ---------------- thin GEMM: q_score = q @ Wqa + bqa  (N = 16 heads) ----------------
__global__ void __launch_bounds__(256)
qscore_wmma(const unsigned short* __restrict__ Qbf,
            const unsigned short* __restrict__ WqaT,
            const float* __restrict__ bqa,
            float* __restrict__ qscore)
{
  const int tid = threadIdx.x, lane = tid & 31, wave = tid >> 5;
  const int rowBase = blockIdx.x * 128 + wave * 16;
  const int r = lane & 15, half = lane >> 4;
  v8f acc;
  #pragma unroll
  for (int e = 0; e < 8; ++e) acc[e] = 0.f;
  const unsigned short* arow = Qbf  + (size_t)(rowBase + r) * D_DIM;
  const unsigned short* bcol = WqaT + (size_t)r * D_DIM;
  for (int k0 = 0; k0 < D_DIM; k0 += 32) {
    Frag16 a, b;
    a.q[0] = *(const uint4*)(arow + k0 + half * 8);
    a.q[1] = *(const uint4*)(arow + k0 + half * 8 + 16);
    b.q[0] = *(const uint4*)(bcol + k0 + half * 8);
    b.q[1] = *(const uint4*)(bcol + k0 + half * 8 + 16);
    acc = __builtin_amdgcn_wmma_f32_16x16x32_bf16(
        false, a.v, false, b.v, (short)0, acc, false, false);
  }
  float bv = bqa[r];
  #pragma unroll
  for (int v = 0; v < 8; ++v)
    qscore[(size_t)(rowBase + half * 8 + v) * H_NUM + r] = acc[v] + bv;
}

// ---------------- fused per-(b,h): softmax -> pooled_q -> qk softmax -> pooled_k ----------------
__global__ void __launch_bounds__(256)
pooled_attention(const float* __restrict__ qscore,  // [ROWS,H]
                 const float* __restrict__ Q,       // [ROWS,D] f32
                 const float* __restrict__ K,       // [ROWS,D] f32
                 const float* __restrict__ mask,    // [B,S]
                 float* __restrict__ pooledK)       // [B,D]
{
  __shared__ float sc[S_LEN];
  __shared__ float red[256];
  __shared__ float pq[DH];
  const int tid = threadIdx.x;
  const int b = blockIdx.x >> 4, h = blockIdx.x & 15;
  const float scale = 0.125f;  // 1/sqrt(DH)
  const size_t rowOff = (size_t)b * S_LEN;

  // ---- phase 1: softmax(q_score) ----
  float lmax = -3.4e38f;
  for (int s = tid; s < S_LEN; s += 256) {
    float x = qscore[(rowOff + s) * H_NUM + h] * scale + mask[rowOff + s];
    sc[s] = x; lmax = fmaxf(lmax, x);
  }
  red[tid] = lmax; __syncthreads();
  for (int off = 128; off > 0; off >>= 1) {
    if (tid < off) red[tid] = fmaxf(red[tid], red[tid + off]);
    __syncthreads();
  }
  float mx = red[0]; __syncthreads();
  float lsum = 0.f;
  for (int s = tid; s < S_LEN; s += 256) { float e = __expf(sc[s] - mx); sc[s] = e; lsum += e; }
  red[tid] = lsum; __syncthreads();
  for (int off = 128; off > 0; off >>= 1) {
    if (tid < off) red[tid] += red[tid + off];
    __syncthreads();
  }
  float inv = 1.f / red[0]; __syncthreads();

  // pooled_q[dh] = sum_s w[s] * q[b,s,h*DH+dh]
  {
    const int dh = tid & 63, g = tid >> 6;
    float a = 0.f;
    for (int s = g; s < S_LEN; s += 4)
      a += sc[s] * Q[(rowOff + s) * D_DIM + h * DH + dh];
    red[tid] = a; __syncthreads();
    if (g == 0) pq[dh] = (red[dh] + red[64 + dh] + red[128 + dh] + red[192 + dh]) * inv;
    __syncthreads();
  }

  // ---- phase 2: qk_score = k . pooled_q, softmax, pooled_k ----
  lmax = -3.4e38f;
  for (int s = tid; s < S_LEN; s += 256) {
    const float* kr = K + (rowOff + s) * D_DIM + h * DH;
    float dot = 0.f;
    #pragma unroll
    for (int d = 0; d < DH; d += 4) {
      float4 kv = *(const float4*)(kr + d);
      dot += kv.x * pq[d] + kv.y * pq[d+1] + kv.z * pq[d+2] + kv.w * pq[d+3];
    }
    float x = dot * scale + mask[rowOff + s];
    sc[s] = x; lmax = fmaxf(lmax, x);
  }
  red[tid] = lmax; __syncthreads();
  for (int off = 128; off > 0; off >>= 1) {
    if (tid < off) red[tid] = fmaxf(red[tid], red[tid + off]);
    __syncthreads();
  }
  mx = red[0]; __syncthreads();
  lsum = 0.f;
  for (int s = tid; s < S_LEN; s += 256) { float e = __expf(sc[s] - mx); sc[s] = e; lsum += e; }
  red[tid] = lsum; __syncthreads();
  for (int off = 128; off > 0; off >>= 1) {
    if (tid < off) red[tid] += red[tid + off];
    __syncthreads();
  }
  inv = 1.f / red[0]; __syncthreads();
  {
    const int dh = tid & 63, g = tid >> 6;
    float a = 0.f;
    for (int s = g; s < S_LEN; s += 4)
      a += sc[s] * K[(rowOff + s) * D_DIM + h * DH + dh];
    red[tid] = a; __syncthreads();
    if (g == 0)
      pooledK[(size_t)b * D_DIM + h * DH + dh] =
          (red[dh] + red[64 + dh] + red[128 + dh] + red[192 + dh]) * inv;
  }
}

// ---------------- host launcher ----------------
extern "C" void kernel_launch(void* const* d_in, const int* in_sizes, int n_in,
                              void* d_out, int out_size, void* d_ws, size_t ws_size,
                              hipStream_t stream) {
  (void)in_sizes; (void)n_in; (void)out_size; (void)ws_size;
  const float* hs   = (const float*)d_in[0];
  const float* mask = (const float*)d_in[1];
  const float* Wq   = (const float*)d_in[2];
  const float* bq   = (const float*)d_in[3];
  const float* Wqa  = (const float*)d_in[4];
  const float* bqa  = (const float*)d_in[5];
  const float* Wk   = (const float*)d_in[6];
  const float* bk   = (const float*)d_in[7];
  const float* Wt   = (const float*)d_in[8];
  const float* bt   = (const float*)d_in[9];
  float* out = (float*)d_out;

  char* ws = (char*)d_ws;
  size_t off = 0;
  auto carve = [&](size_t bytes) -> void* {
    void* p = ws + off;
    off = (off + bytes + 255) & ~(size_t)255;
    return p;
  };
  const size_t actHalf = (size_t)ROWS * D_DIM * sizeof(unsigned short);
  const size_t actF32  = (size_t)ROWS * D_DIM * sizeof(float);
  const size_t wHalf   = (size_t)D_DIM * D_DIM * sizeof(unsigned short);

  unsigned short* hs_bf  = (unsigned short*)carve(actHalf);
  unsigned short* q_bf   = (unsigned short*)carve(actHalf);
  unsigned short* wgt_bf = (unsigned short*)carve(actHalf);
  unsigned short* WqT    = (unsigned short*)carve(wHalf);
  unsigned short* WkT    = (unsigned short*)carve(wHalf);
  unsigned short* WtT    = (unsigned short*)carve(wHalf);
  unsigned short* WqaT   = (unsigned short*)carve((size_t)H_NUM * D_DIM * sizeof(unsigned short));
  float* q_f32   = (float*)carve(actF32);
  float* k_f32   = (float*)carve(actF32);
  float* qscore  = (float*)carve((size_t)ROWS * H_NUM * sizeof(float));
  float* pooledK = (float*)carve((size_t)B_NUM * D_DIM * sizeof(float));

  const int nAct = ROWS * D_DIM;       // 16,777,216

  // 1) pack: activations row-major, weights transposed (n-major)
  pack_bf16<<<nAct / 1024, 256, 0, stream>>>(hs, hs_bf, nAct);
  dim3 tBlk(32, 8), tGrd(D_DIM / 32, D_DIM / 32);
  pack_transpose<<<tGrd, tBlk, 0, stream>>>(Wq, WqT, D_DIM, D_DIM);
  pack_transpose<<<tGrd, tBlk, 0, stream>>>(Wk, WkT, D_DIM, D_DIM);
  pack_transpose<<<tGrd, tBlk, 0, stream>>>(Wt, WtT, D_DIM, D_DIM);
  pack_wqaT<<<(D_DIM * H_NUM) / 256, 256, 0, stream>>>(Wqa, WqaT);

  dim3 gGemm(D_DIM / 256, ROWS / 128);  // (4, 128)

  // 2) q = hs @ Wq + bq  (f32 + bf16 copy)
  gemm_bf16_wmma<<<gGemm, 256, 0, stream>>>(hs_bf, WqT, bq, nullptr, q_f32, q_bf);
  // 3) k = hs @ Wk + bk
  gemm_bf16_wmma<<<gGemm, 256, 0, stream>>>(hs_bf, WkT, bk, nullptr, k_f32, nullptr);
  // 4) q_score = q @ Wqa + bqa
  qscore_wmma<<<ROWS / 128, 256, 0, stream>>>(q_bf, WqaT, bqa, qscore);
  // 5) fused softmax -> pooled_q -> qk softmax -> pooled_k
  pooled_attention<<<B_NUM * H_NUM, 256, 0, stream>>>(qscore, q_f32, k_f32, mask, pooledK);
  // 6) weighted = pooled_k ⊙ q  (bf16)
  scale_q<<<nAct / 1024, 256, 0, stream>>>(q_f32, pooledK, wgt_bf);
  // 7) out = weighted @ Wt + bt + q
  gemm_bf16_wmma<<<gGemm, 256, 0, stream>>>(wgt_bf, WtT, bt, q_f32, out, nullptr);
}
